// GatedGCNNet_32272384262278
// MI455X (gfx1250) — matrix operand order repaired and checked by reading, not compiled, not run on hardware
//
#include <hip/hip_runtime.h>

// ---------------------------------------------------------------------------
// GatedGCN-LSPE forward for MI455X (gfx1250, wave32, WMMA).
// All dense linears run through v_wmma_f32_16x16x32_bf16 (f32 accumulate).
// Edge gather/scatter uses float4 VMEM + global_atomic_add_f32.
// Workspace requirement: ~884 MB (2x E*D + 10x N*D f32 buffers + small).
// ---------------------------------------------------------------------------

#define N_NODES     40960
#define N_EDGES     655360
#define DIM         128
#define POS_DIM     16
#define N_GRAPHS    64
#define NODES_PER_G 640
#define N_LAYERS    4
#define EPS_GATE    1e-6f
#define BN_EPS      1e-5f

#define CEIL(a, b) (((a) + (b) - 1) / (b))

typedef __attribute__((ext_vector_type(16))) __bf16 v16bf;
typedef __attribute__((ext_vector_type(8)))  float  v8f;

union Frag16 { v16bf v; unsigned int u[8]; };

__device__ __forceinline__ unsigned short f2bf(float f) {
  unsigned int x = __float_as_uint(f);
  x += 0x7fffu + ((x >> 16) & 1u);           // round-to-nearest-even
  return (unsigned short)(x >> 16);
}
__device__ __forceinline__ unsigned int pack2bf(float lo, float hi) {
  return (unsigned int)f2bf(lo) | ((unsigned int)f2bf(hi) << 16);
}
__device__ __forceinline__ float sigmoidf_(float x) {
  return 1.0f / (1.0f + expf(-x));
}

#define GEMM_FLAG_BIAS 1
#define GEMM_FLAG_RELU 2
#define GEMM_FLAG_ACC  4

// ---------------------------------------------------------------------------
// Generic tiled WMMA GEMM: C[M,N] = A[M,K] * W[K,N] (+bias) (+C) (relu)
// 256 threads = 8 wave32 waves; 128x128 output tile; K stepped by 32.
// LDS tiles hold bf16 K-pairs packed in uint32, laid out to match the
// CDNA5 WMMA VGPR fragment layouts (ISA 7.12.2).
// FULL=true: tile-aligned fast path (M%128==0, K%32==0, N%128==0) with no
// bounds predication and global_prefetch of the next A K-tile.
// ---------------------------------------------------------------------------
template <bool FULL>
__global__ __launch_bounds__(256)
void wmma_gemm_kernel(const float* __restrict__ A, int lda,
                      const float* __restrict__ W, int ldw,
                      const float* __restrict__ bias,
                      float* __restrict__ C, int ldc,
                      int M, int K, int N, int flags)
{
  __shared__ unsigned int sA[128][16];   // [m][kpair] bf16x2
  __shared__ unsigned int sB[128][16];   // [n][kpair] bf16x2 (W transposed)

  const int tid  = threadIdx.x;
  const int wave = tid >> 5;
  const int lane = tid & 31;
  const int ln   = lane & 15;
  const int lhi  = lane >> 4;            // 0 or 1
  const int m0   = blockIdx.x * 128;
  const int n0   = blockIdx.y * 128;
  const int wm   = wave * 16;            // this wave's row strip

  v8f acc[8];
#pragma unroll
  for (int i = 0; i < 8; ++i)
#pragma unroll
    for (int e = 0; e < 8; ++e) acc[i][e] = 0.0f;

  for (int k0 = 0; k0 < K; k0 += 32) {
    // ---- stage A tile (128 rows x 32 cols) as bf16 pairs ----
    {
      const int colBase = (tid & 7) * 4;
      const int rowBase = tid >> 3;
#pragma unroll
      for (int rep = 0; rep < 4; ++rep) {
        const int r  = rowBase + rep * 32;
        const int gm = m0 + r;
        const int gk = k0 + colBase;
        float x0 = 0.f, x1 = 0.f, x2 = 0.f, x3 = 0.f;
        if (FULL) {
          const float* p = A + (size_t)gm * lda + gk;
          float4 v = *(const float4*)p;
          x0 = v.x; x1 = v.y; x2 = v.z; x3 = v.w;
          if (k0 + 32 < K)                       // prefetch next K-tile
            __builtin_prefetch(p + 32, 0, 0);
        } else if (gm < M) {
          const float* p = A + (size_t)gm * lda + gk;
          if (gk + 3 < K) {
            float4 v = *(const float4*)p;
            x0 = v.x; x1 = v.y; x2 = v.z; x3 = v.w;
          } else {
            if (gk + 0 < K) x0 = p[0];
            if (gk + 1 < K) x1 = p[1];
            if (gk + 2 < K) x2 = p[2];
            if (gk + 3 < K) x3 = p[3];
          }
        }
        sA[r][(colBase >> 1) + 0] = pack2bf(x0, x1);
        sA[r][(colBase >> 1) + 1] = pack2bf(x2, x3);
      }
    }
    // ---- stage W tile (32 x 128) transposed to [n][k] bf16 pairs ----
    {
      const int n  = tid & 127;
      const int kh = (tid >> 7) * 16;    // 0 or 16
      const int gn = n0 + n;
#pragma unroll
      for (int j = 0; j < 8; ++j) {
        const int k   = kh + 2 * j;
        const int gk0 = k0 + k;
        float lo = 0.f, hi = 0.f;
        if (FULL) {
          lo = W[(size_t)(gk0 + 0) * ldw + gn];
          hi = W[(size_t)(gk0 + 1) * ldw + gn];
        } else if (gn < N) {
          if (gk0 + 0 < K) lo = W[(size_t)(gk0 + 0) * ldw + gn];
          if (gk0 + 1 < K) hi = W[(size_t)(gk0 + 1) * ldw + gn];
        }
        sB[n][k >> 1] = pack2bf(lo, hi);
      }
    }
    __syncthreads();

    // ---- A fragment: 16-bit A 16x32 layout (K+8 swap for lanes 16-31) ----
    Frag16 fa;
#pragma unroll
    for (int j = 0; j < 8; ++j) {
      const int idx = j + ((j >> 2) << 2) + lhi * 4;  // kpair index
      fa.u[j] = sA[wm + ln][idx];
    }
    // ---- preload all 8 B fragments, then issue WMMAs back-to-back ----
    Frag16 fb[8];
#pragma unroll
    for (int nt = 0; nt < 8; ++nt)
#pragma unroll
      for (int j = 0; j < 8; ++j)
        fb[nt].u[j] = sB[nt * 16 + ln][j + ((j >> 2) << 2) + lhi * 4];
#pragma unroll
    for (int nt = 0; nt < 8; ++nt)
      acc[nt] = __builtin_amdgcn_wmma_f32_16x16x32_bf16(
          false, fa.v, false, fb[nt].v, (short)0, acc[nt], false, false);
    __syncthreads();
  }

  // ---- epilogue: C/D 16x16 f32 layout (VGPR r -> M=r / r+8) ----
#pragma unroll
  for (int nt = 0; nt < 8; ++nt) {
    const int n = n0 + nt * 16 + ln;
    if (!FULL && n >= N) continue;
    const float bv = (flags & GEMM_FLAG_BIAS) ? bias[n] : 0.0f;
#pragma unroll
    for (int r = 0; r < 8; ++r) {
      const int m = m0 + wm + r + lhi * 8;
      if (!FULL && m >= M) continue;
      float val = acc[nt][r] + bv;
      const size_t idx = (size_t)m * ldc + n;
      if (flags & GEMM_FLAG_ACC)  val += C[idx];
      if (flags & GEMM_FLAG_RELU) val = fmaxf(val, 0.0f);
      C[idx] = val;
    }
  }
}

// ---------------------------------------------------------------------------
// Embedding gather: out[r,:] = table[idx[r],:]   (D columns)
// ---------------------------------------------------------------------------
__global__ void gather_rows_kernel(const float* __restrict__ table,
                                   const int* __restrict__ idx,
                                   float* __restrict__ out, int rows)
{
  int i = blockIdx.x * blockDim.x + threadIdx.x;
  if (i >= rows * DIM) return;
  int r = i >> 7, c = i & 127;
  out[i] = table[idx[r] * DIM + c];
}

// ---------------------------------------------------------------------------
// hat += B1h[src] + B2h[dst]; sigsum[dst] += sigmoid(hat)   (float4/thread)
// ---------------------------------------------------------------------------
__global__ void edge_gate_kernel(float* __restrict__ hat,
                                 const float* __restrict__ b1h,
                                 const float* __restrict__ b2h,
                                 const int* __restrict__ src,
                                 const int* __restrict__ dst,
                                 float* __restrict__ sigsum)
{
  int gid = blockIdx.x * blockDim.x + threadIdx.x;
  if (gid >= N_EDGES * 32) return;
  int e = gid >> 5;
  int q = (gid & 31) << 2;
  int s = src[e], d = dst[e];
  float4 hv = *(float4*)(hat + (size_t)e * DIM + q);
  float4 a  = *(const float4*)(b1h + (size_t)s * DIM + q);
  float4 b  = *(const float4*)(b2h + (size_t)d * DIM + q);
  hv.x += a.x + b.x; hv.y += a.y + b.y;
  hv.z += a.z + b.z; hv.w += a.w + b.w;
  *(float4*)(hat + (size_t)e * DIM + q) = hv;
  float* ss = sigsum + (size_t)d * DIM + q;
  atomicAdd(ss + 0, sigmoidf_(hv.x));
  atomicAdd(ss + 1, sigmoidf_(hv.y));
  atomicAdd(ss + 2, sigmoidf_(hv.z));
  atomicAdd(ss + 3, sigmoidf_(hv.w));
}

// ---------------------------------------------------------------------------
// eta = sigmoid(hat)/(sigsum[dst]+eps);
// hacc[dst] += eta * a2hp[src]; pacc[dst] += eta * c2p[src]
// ---------------------------------------------------------------------------
__global__ void edge_aggregate_kernel(const float* __restrict__ hat,
                                      const float* __restrict__ sigsum,
                                      const float* __restrict__ a2hp,
                                      const float* __restrict__ c2p,
                                      const int* __restrict__ src,
                                      const int* __restrict__ dst,
                                      float* __restrict__ hacc,
                                      float* __restrict__ pacc)
{
  int gid = blockIdx.x * blockDim.x + threadIdx.x;
  if (gid >= N_EDGES * 32) return;
  int e = gid >> 5;
  int q = (gid & 31) << 2;
  int s = src[e], d = dst[e];
  float4 hv = *(const float4*)(hat    + (size_t)e * DIM + q);
  float4 ss = *(const float4*)(sigsum + (size_t)d * DIM + q);
  float4 av = *(const float4*)(a2hp   + (size_t)s * DIM + q);
  float4 cv = *(const float4*)(c2p    + (size_t)s * DIM + q);
  float e0 = sigmoidf_(hv.x) / (ss.x + EPS_GATE);
  float e1 = sigmoidf_(hv.y) / (ss.y + EPS_GATE);
  float e2 = sigmoidf_(hv.z) / (ss.z + EPS_GATE);
  float e3 = sigmoidf_(hv.w) / (ss.w + EPS_GATE);
  float* hd = hacc + (size_t)d * DIM + q;
  float* pd = pacc + (size_t)d * DIM + q;
  atomicAdd(hd + 0, e0 * av.x); atomicAdd(hd + 1, e1 * av.y);
  atomicAdd(hd + 2, e2 * av.z); atomicAdd(hd + 3, e3 * av.w);
  atomicAdd(pd + 0, e0 * cv.x); atomicAdd(pd + 1, e1 * cv.y);
  atomicAdd(pd + 2, e2 * cv.z); atomicAdd(pd + 3, e3 * cv.w);
}

// ---------------------------------------------------------------------------
// Per-column sum / sumsq over rows (BatchNorm stats). 128 threads/block.
// ---------------------------------------------------------------------------
__global__ void col_stats_kernel(const float* __restrict__ X, int rows,
                                 float* __restrict__ stats)
{
  const int c = threadIdx.x;                 // column 0..127
  int r0 = blockIdx.x * 512;
  int r1 = r0 + 512; if (r1 > rows) r1 = rows;
  float s = 0.f, sq = 0.f;
  for (int r = r0; r < r1; ++r) {
    float v = X[(size_t)r * DIM + c];
    s += v; sq += v * v;
  }
  atomicAdd(&stats[c], s);
  atomicAdd(&stats[DIM + c], sq);
}

// base += relu( (x - mean) * rsqrt(var + eps) * g + b )
__global__ void bn_relu_residual_kernel(float* __restrict__ base,
                                        const float* __restrict__ x,
                                        const float* __restrict__ stats,
                                        const float* __restrict__ g,
                                        const float* __restrict__ b,
                                        float invM, int total)
{
  int i = blockIdx.x * blockDim.x + threadIdx.x;
  if (i >= total) return;
  int c = i & 127;
  float m = stats[c] * invM;
  float v = stats[DIM + c] * invM - m * m;   // biased variance (var(0))
  float y = (x[i] - m) * rsqrtf(v + BN_EPS) * g[c] + b[c];
  base[i] += fmaxf(y, 0.0f);
}

__global__ void tanh_residual_kernel(float* __restrict__ base,
                                     const float* __restrict__ x, int total)
{
  int i = blockIdx.x * blockDim.x + threadIdx.x;
  if (i >= total) return;
  base[i] += tanhf(x[i]);
}

// ---- LSPE post-processing over po [N, POS_DIM] ----
__global__ void seg_sum_po_kernel(const float* __restrict__ po,
                                  const int* __restrict__ gidx,
                                  float* __restrict__ gsum)
{
  int i = blockIdx.x * blockDim.x + threadIdx.x;
  if (i >= N_NODES * POS_DIM) return;
  int n = i >> 4, j = i & 15;
  atomicAdd(&gsum[gidx[n] * POS_DIM + j], po[i]);
}

__global__ void center_sq_po_kernel(float* __restrict__ po,
                                    const int* __restrict__ gidx,
                                    const float* __restrict__ gsum,
                                    float* __restrict__ gsq)
{
  int i = blockIdx.x * blockDim.x + threadIdx.x;
  if (i >= N_NODES * POS_DIM) return;
  int n = i >> 4, j = i & 15;
  int g = gidx[n];
  float v = po[i] - gsum[g * POS_DIM + j] * (1.0f / NODES_PER_G);
  po[i] = v;
  atomicAdd(&gsq[g * POS_DIM + j], v * v);
}

__global__ void norm_po_kernel(float* __restrict__ po,
                               const int* __restrict__ gidx,
                               const float* __restrict__ gsq)
{
  int i = blockIdx.x * blockDim.x + threadIdx.x;
  if (i >= N_NODES * POS_DIM) return;
  int n = i >> 4, j = i & 15;
  po[i] = po[i] / sqrtf(gsq[gidx[n] * POS_DIM + j]);
}

// Mean readout: hg[g,:] += hp2[n,:] / 640
__global__ void readout_mean_kernel(const float* __restrict__ hp2,
                                    const int* __restrict__ gidx,
                                    float* __restrict__ hg)
{
  int i = blockIdx.x * blockDim.x + threadIdx.x;
  if (i >= N_NODES * DIM) return;
  int n = i >> 7, c = i & 127;
  atomicAdd(&hg[gidx[n] * DIM + c], hp2[i] * (1.0f / NODES_PER_G));
}

// ---------------------------------------------------------------------------
static inline void gemm(hipStream_t s, const float* A, int lda,
                        const float* W, int ldw, const float* bias,
                        float* C, int ldc, int M, int K, int N, int flags)
{
  dim3 grid(CEIL(M, 128), CEIL(N, 128));
  const bool full = (M % 128 == 0) && (K % 32 == 0) && (N % 128 == 0);
  if (full)
    wmma_gemm_kernel<true><<<grid, 256, 0, s>>>(A, lda, W, ldw, bias, C, ldc,
                                                M, K, N, flags);
  else
    wmma_gemm_kernel<false><<<grid, 256, 0, s>>>(A, lda, W, ldw, bias, C, ldc,
                                                 M, K, N, flags);
}

extern "C" void kernel_launch(void* const* d_in, const int* in_sizes, int n_in,
                              void* d_out, int out_size, void* d_ws, size_t ws_size,
                              hipStream_t stream)
{
  (void)in_sizes; (void)n_in; (void)out_size; (void)ws_size;

  const int*   h_atoms = (const int*)d_in[0];
  const int*   e_bonds = (const int*)d_in[1];
  const int*   src     = (const int*)d_in[2];
  const int*   dst     = (const int*)d_in[3];
  const int*   ngid    = (const int*)d_in[4];
  const float* p       = (const float*)d_in[5];
  const float* W_emb_h = (const float*)d_in[6];
  const float* W_emb_p = (const float*)d_in[7];
  const float* b_emb_p = (const float*)d_in[8];
  const float* W_emb_e = (const float*)d_in[9];
  const float* A1_w = (const float*)d_in[10];
  const float* A1_b = (const float*)d_in[11];
  const float* A2_w = (const float*)d_in[12];
  const float* A2_b = (const float*)d_in[13];
  const float* B1_w = (const float*)d_in[14];
  const float* B1_b = (const float*)d_in[15];
  const float* B2_w = (const float*)d_in[16];
  const float* B2_b = (const float*)d_in[17];
  const float* B3_w = (const float*)d_in[18];
  const float* B3_b = (const float*)d_in[19];
  const float* C1_w = (const float*)d_in[20];
  const float* C1_b = (const float*)d_in[21];
  const float* C2_w = (const float*)d_in[22];
  const float* C2_b = (const float*)d_in[23];
  const float* bn_h_g = (const float*)d_in[24];
  const float* bn_h_b = (const float*)d_in[25];
  const float* bn_e_g = (const float*)d_in[26];
  const float* bn_e_b = (const float*)d_in[27];
  const float* p_out_w = (const float*)d_in[28];
  const float* p_out_b = (const float*)d_in[29];
  const float* Whp_w   = (const float*)d_in[30];
  const float* Whp_b   = (const float*)d_in[31];
  const float* mlp_w0 = (const float*)d_in[32];
  const float* mlp_b0 = (const float*)d_in[33];
  const float* mlp_w1 = (const float*)d_in[34];
  const float* mlp_b1 = (const float*)d_in[35];
  const float* mlp_w2 = (const float*)d_in[36];
  const float* mlp_b2 = (const float*)d_in[37];

  // ---- workspace carve-up ----
  float* ws = (float*)d_ws;
  size_t off = 0;
  auto alloc = [&](size_t n) { float* r = ws + off; off += n; return r; };
  float* h      = alloc((size_t)N_NODES * DIM);
  float* pc     = alloc((size_t)N_NODES * DIM);
  float* ef     = alloc((size_t)N_EDGES * DIM);
  float* hat    = alloc((size_t)N_EDGES * DIM);
  float* b1h    = alloc((size_t)N_NODES * DIM);
  float* b2h    = alloc((size_t)N_NODES * DIM);
  float* sigsum = alloc((size_t)N_NODES * DIM);
  float* hacc   = alloc((size_t)N_NODES * DIM);
  float* pacc   = alloc((size_t)N_NODES * DIM);
  float* a2hp   = alloc((size_t)N_NODES * DIM);
  float* c2p    = alloc((size_t)N_NODES * DIM);
  float* hp2    = alloc((size_t)N_NODES * DIM);
  float* po     = alloc((size_t)N_NODES * POS_DIM);
  float* hg     = alloc((size_t)N_GRAPHS * DIM);
  float* y0     = alloc(64 * 64);
  float* y1     = alloc(64 * 32);
  float* statsH = alloc(2 * DIM);
  float* statsE = alloc(2 * DIM);
  float* gsum   = alloc(N_GRAPHS * POS_DIM);
  float* gsq    = alloc(N_GRAPHS * POS_DIM);

  const int T = 256;

  // ---- embeddings ----
  gather_rows_kernel<<<CEIL(N_NODES * DIM, T), T, 0, stream>>>(W_emb_h, h_atoms, h, N_NODES);
  gemm(stream, p, POS_DIM, W_emb_p, DIM, b_emb_p, pc, DIM,
       N_NODES, POS_DIM, DIM, GEMM_FLAG_BIAS);
  gather_rows_kernel<<<CEIL(N_EDGES * DIM, T), T, 0, stream>>>(W_emb_e, e_bonds, ef, N_EDGES);

  // ---- GatedGCN-LSPE layers ----
  for (int l = 0; l < N_LAYERS; ++l) {
    const float* a1w = A1_w + (size_t)l * 2 * DIM * DIM;
    const float* a2w = A2_w + (size_t)l * 2 * DIM * DIM;
    const float* b1w = B1_w + (size_t)l * DIM * DIM;
    const float* b2w = B2_w + (size_t)l * DIM * DIM;
    const float* b3w = B3_w + (size_t)l * DIM * DIM;
    const float* c1w = C1_w + (size_t)l * DIM * DIM;
    const float* c2w = C2_w + (size_t)l * DIM * DIM;

    // node/edge linears for the gate
    gemm(stream, h,  DIM, b1w, DIM, B1_b + l * DIM, b1h, DIM, N_NODES, DIM, DIM, GEMM_FLAG_BIAS);
    gemm(stream, h,  DIM, b2w, DIM, B2_b + l * DIM, b2h, DIM, N_NODES, DIM, DIM, GEMM_FLAG_BIAS);
    gemm(stream, ef, DIM, b3w, DIM, B3_b + l * DIM, hat, DIM, N_EDGES, DIM, DIM, GEMM_FLAG_BIAS);

    hipMemsetAsync(sigsum, 0, (size_t)N_NODES * DIM * sizeof(float), stream);
    edge_gate_kernel<<<CEIL(N_EDGES * 32, T), T, 0, stream>>>(hat, b1h, b2h, src, dst, sigsum);

    // hp = [h, pc]: split 2D->D GEMMs with accumulate
    gemm(stream, h,  DIM, a1w,             DIM, A1_b + l * DIM, hacc, DIM, N_NODES, DIM, DIM, GEMM_FLAG_BIAS);
    gemm(stream, pc, DIM, a1w + DIM * DIM, DIM, nullptr,        hacc, DIM, N_NODES, DIM, DIM, GEMM_FLAG_ACC);
    gemm(stream, h,  DIM, a2w,             DIM, A2_b + l * DIM, a2hp, DIM, N_NODES, DIM, DIM, GEMM_FLAG_BIAS);
    gemm(stream, pc, DIM, a2w + DIM * DIM, DIM, nullptr,        a2hp, DIM, N_NODES, DIM, DIM, GEMM_FLAG_ACC);
    gemm(stream, pc, DIM, c1w, DIM, C1_b + l * DIM, pacc, DIM, N_NODES, DIM, DIM, GEMM_FLAG_BIAS);
    gemm(stream, pc, DIM, c2w, DIM, C2_b + l * DIM, c2p,  DIM, N_NODES, DIM, DIM, GEMM_FLAG_BIAS);

    edge_aggregate_kernel<<<CEIL(N_EDGES * 32, T), T, 0, stream>>>(
        hat, sigsum, a2hp, c2p, src, dst, hacc, pacc);

    // BN(h_new) + relu + residual into h
    hipMemsetAsync(statsH, 0, 2 * DIM * sizeof(float), stream);
    col_stats_kernel<<<CEIL(N_NODES, 512), DIM, 0, stream>>>(hacc, N_NODES, statsH);
    bn_relu_residual_kernel<<<CEIL(N_NODES * DIM, T), T, 0, stream>>>(
        h, hacc, statsH, bn_h_g + l * DIM, bn_h_b + l * DIM,
        1.0f / N_NODES, N_NODES * DIM);

    // BN(hat_eta) + relu + residual into ef
    hipMemsetAsync(statsE, 0, 2 * DIM * sizeof(float), stream);
    col_stats_kernel<<<CEIL(N_EDGES, 512), DIM, 0, stream>>>(hat, N_EDGES, statsE);
    bn_relu_residual_kernel<<<CEIL(N_EDGES * DIM, T), T, 0, stream>>>(
        ef, hat, statsE, bn_e_g + l * DIM, bn_e_b + l * DIM,
        1.0f / N_EDGES, N_EDGES * DIM);

    // pc += tanh(p_new)
    tanh_residual_kernel<<<CEIL(N_NODES * DIM, T), T, 0, stream>>>(pc, pacc, N_NODES * DIM);
  }

  // ---- LSPE post-processing ----
  gemm(stream, pc, DIM, p_out_w, POS_DIM, p_out_b, po, POS_DIM,
       N_NODES, DIM, POS_DIM, GEMM_FLAG_BIAS);
  hipMemsetAsync(gsum, 0, N_GRAPHS * POS_DIM * sizeof(float), stream);
  seg_sum_po_kernel<<<CEIL(N_NODES * POS_DIM, T), T, 0, stream>>>(po, ngid, gsum);
  hipMemsetAsync(gsq, 0, N_GRAPHS * POS_DIM * sizeof(float), stream);
  center_sq_po_kernel<<<CEIL(N_NODES * POS_DIM, T), T, 0, stream>>>(po, ngid, gsum, gsq);
  norm_po_kernel<<<CEIL(N_NODES * POS_DIM, T), T, 0, stream>>>(po, ngid, gsq);

  // hp2 = [h, po] @ Whp + b  (split K = 128 + 16)
  gemm(stream, h,  DIM,     Whp_w,                     DIM, Whp_b,  hp2, DIM, N_NODES, DIM,     DIM, GEMM_FLAG_BIAS);
  gemm(stream, po, POS_DIM, Whp_w + (size_t)DIM * DIM, DIM, nullptr, hp2, DIM, N_NODES, POS_DIM, DIM, GEMM_FLAG_ACC);

  // mean readout + MLP head
  hipMemsetAsync(hg, 0, N_GRAPHS * DIM * sizeof(float), stream);
  readout_mean_kernel<<<CEIL(N_NODES * DIM, T), T, 0, stream>>>(hp2, ngid, hg);

  gemm(stream, hg, DIM, mlp_w0, 64, mlp_b0, y0, 64, N_GRAPHS, DIM, 64,
       GEMM_FLAG_BIAS | GEMM_FLAG_RELU);
  gemm(stream, y0, 64, mlp_w1, 32, mlp_b1, y1, 32, N_GRAPHS, 64, 32,
       GEMM_FLAG_BIAS | GEMM_FLAG_RELU);
  gemm(stream, y1, 32, mlp_w2, 1, mlp_b2, (float*)d_out, 1, N_GRAPHS, 32, 1,
       GEMM_FLAG_BIAS);
}
